// SentenceEncoder_88699664597345
// MI455X (gfx1250) — matrix operand order repaired and checked by reading, not compiled, not run on hardware
//
#include <hip/hip_runtime.h>
#include <hip/hip_bf16.h>
#include <cstdint>

// ---------------- problem constants ----------------
#define BATCH 256
#define TSEQ  256
#define DIN   300
#define HID   200
#define H3    600
#define OUTN  100

#define NPAD   608   // 38 tiles of 16 (>= 600)
#define NTILES 38
#define KD_PAD 320   // 10 K-chunks of 32 (>= 300)
#define KD_CH  10
#define KH_PAD 224   // 7 K-chunks of 32 (>= 200)
#define KH_CH  7

#define AS1 __attribute__((address_space(1)))
#define AS3 __attribute__((address_space(3)))

typedef __attribute__((ext_vector_type(16))) __bf16 v16bf;
typedef __attribute__((ext_vector_type(8)))  __bf16 v8bf;
typedef __attribute__((ext_vector_type(8)))  float  v8f;
typedef int b128vec __attribute__((vector_size(16)));   // int4 as in builtin sig

// A-operand fragment for v_wmma_f32_16x16x32_bf16 (ISA 7.12.2):
// lane m (m=lane&15), half=lane>>4: first 8 elems = K[half*8 .. +7],
// second 8 elems = K[16+half*8 .. +7]  ->  two 16B loads at p and p+16.
__device__ __forceinline__ v16bf load_a_frag(const __bf16* p) {
  v8bf lo = *(const v8bf*)(p);
  v8bf hi = *(const v8bf*)(p + 16);
  return __builtin_shufflevector(lo, hi,
      0,1,2,3,4,5,6,7,8,9,10,11,12,13,14,15);
}

// gfx1250 has V_TANH_F32 (TRANS32) — one trans op per gate.
__device__ __forceinline__ float fast_tanh(float x) {
#if __has_builtin(__builtin_amdgcn_tanhf)
  return __builtin_amdgcn_tanhf(x);
#elif __has_builtin(__builtin_amdgcn_tanh_f32)
  return __builtin_amdgcn_tanh_f32(x);
#else
  return tanhf(x);
#endif
}

__device__ __forceinline__ float sigmoidf_(float x) {
#if __has_builtin(__builtin_amdgcn_tanhf) || __has_builtin(__builtin_amdgcn_tanh_f32)
  return 0.5f + 0.5f * fast_tanh(0.5f * x);   // sigma(x) = (1+tanh(x/2))/2
#else
  return 1.0f / (1.0f + __expf(-x));
#endif
}

// ---------------- 1) weight transpose + bf16 convert ----------------
// dst[n*Kpad + k] = (k<K && n<N) ? (bf16)src[k*N + n] : 0
__global__ void transpose_pad_bf16_k(const float* __restrict__ src,
                                     unsigned short* __restrict__ dst_raw,
                                     int K, int N, int Kpad, int Npad) {
  __bf16* dst = (__bf16*)dst_raw;
  int idx = blockIdx.x * blockDim.x + threadIdx.x;
  int total = Kpad * Npad;
  if (idx >= total) return;
  int n = idx / Kpad, k = idx % Kpad;
  float v = (k < K && n < N) ? src[(size_t)k * N + n] : 0.0f;
  dst[idx] = (__bf16)v;
}

// ---------------- 2) x projection: xp = x(@rev) @ Wx + bx ----------------
// grid: (1024 Mtiles of 64 rows, 5 Nblks of 128 cols, 2 dirs), 256 threads.
__global__ void __launch_bounds__(256)
xproj_wmma_k(const float* __restrict__ x, const int* __restrict__ lengths,
             const unsigned short* __restrict__ wxt_raw,  // [2][NPAD][KD_PAD] bf16
             const float* __restrict__ bx_f, const float* __restrict__ bx_b,
             float* __restrict__ xp_f, float* __restrict__ xp_b) {
  __shared__ __bf16 A_s[64 * 32];
  const __bf16* wxt = (const __bf16*)wxt_raw;

  const int tid  = threadIdx.x;
  const int lane = tid & 31, wave = tid >> 5;
  const int half = lane >> 4, lm = lane & 15;
  const int mtile = blockIdx.x;
  const int nblk  = blockIdx.y;
  const int dir   = blockIdx.z;

  const __bf16* wq  = wxt + (size_t)dir * NPAD * KD_PAD;
  float* xp         = dir ? xp_b : xp_f;
  const float* bxp  = dir ? bx_b : bx_f;

  const int col0 = nblk * 128 + wave * 16;
  const bool wactive = (col0 < NPAD);

  // A staging coords: thread loads 8 consecutive k of one row
  const int srow = tid >> 2;          // 0..63
  const int sk   = (tid & 3) * 8;     // 0,8,16,24
  const int gr   = mtile * 64 + srow; // flat row over [B][T]
  const int bb   = gr >> 8;
  const int tt0  = gr & 255;
  int tsrc = tt0;
  if (dir) {
    int L = lengths[bb];
    tsrc = L - 1 - tt0;
    if (tsrc < 0) tsrc = 0;           // clip (reference rev_idx)
  }
  const float* xrow = x + ((size_t)bb * TSEQ + tsrc) * DIN;

  v8f acc[4];
  #pragma unroll
  for (int m = 0; m < 4; ++m) acc[m] = (v8f){};

  for (int kc = 0; kc < KD_CH; ++kc) {
    __syncthreads();
    #pragma unroll
    for (int e = 0; e < 8; ++e) {
      int kg = kc * 32 + sk + e;
      float v = (kg < DIN) ? xrow[kg] : 0.0f;
      A_s[srow * 32 + sk + e] = (__bf16)v;
    }
    __syncthreads();
    if (wactive) {
      v16bf bm = load_a_frag(wq + (size_t)(col0 + lm) * KD_PAD
                                 + kc * 32 + half * 8);
      #pragma unroll
      for (int m = 0; m < 4; ++m) {
        v16bf a = load_a_frag(&A_s[(m * 16 + lm) * 32 + half * 8]);
        acc[m] = __builtin_amdgcn_wmma_f32_16x16x32_bf16(
            false, a, false, bm, (short)0, acc[m], false, false);
      }
    }
  }

  const int col = col0 + lm;
  if (col < H3) {
    float bxs = bxp[col];
    #pragma unroll
    for (int m = 0; m < 4; ++m) {
      #pragma unroll
      for (int g = 0; g < 8; ++g) {
        int rl  = m * 16 + half * 8 + g;         // C layout: M = g + half*8
        int grr = mtile * 64 + rl;
        int b2 = grr >> 8, t2 = grr & 255;
        xp[((size_t)t2 * BATCH + b2) * H3 + col] = acc[m][g] + bxs;
      }
    }
  }
}

// ---------------- 3) GRU scan with LDS-resident Wh^T ----------------
// grid: (16 batch slices, 2 dirs), 256 threads, ~311 KB dynamic LDS.
__global__ void __launch_bounds__(256)
gru_scan_wmma_k(const float* __restrict__ xp_f, const float* __restrict__ xp_b,
                const unsigned short* __restrict__ wht_raw,  // [2][NPAD][KH_PAD] bf16
                const float* __restrict__ bh_f, const float* __restrict__ bh_b,
                const int* __restrict__ lengths,
                float* __restrict__ sums) {                  // [2][B][HID]
  extern __shared__ char smem_raw[];
  __bf16* whts = (__bf16*)smem_raw;              // NPAD*KH_PAD   (272,384 B)
  __bf16* hbf  = whts + NPAD * KH_PAD;           // 16*KH_PAD    (  7,168 B)
  float*  recs = (float*)(hbf + 16 * KH_PAD);    // 16*NPAD      ( 38,912 B)

  const int tid  = threadIdx.x;
  const int lane = tid & 31, wave = tid >> 5;
  const int half = lane >> 4, lm = lane & 15;
  const int slice = blockIdx.x;
  const int dir   = blockIdx.y;
  const int brow0 = slice * 16;

  const float* xp = dir ? xp_b : xp_f;
  const float* bh = dir ? bh_b : bh_f;
  const __bf16* wsrc = (const __bf16*)wht_raw + (size_t)dir * NPAD * KH_PAD;

  // one-time: Wh^T (bf16) -> LDS (gfx1250 async-to-LDS path when available)
#if __has_builtin(__builtin_amdgcn_global_load_async_to_lds_b128)
  for (int i = tid; i < (NPAD * KH_PAD) / 8; i += 256) {
    __builtin_amdgcn_global_load_async_to_lds_b128(
        (AS1 b128vec*)(uintptr_t)(wsrc + (size_t)i * 8),
        (AS3 b128vec*)(uintptr_t)(whts + (size_t)i * 8), 0, 0);
  }
#if __has_builtin(__builtin_amdgcn_s_wait_asynccnt)
  __builtin_amdgcn_s_wait_asynccnt(0);
#else
  asm volatile("s_wait_asynccnt 0" ::: "memory");
#endif
#else
  {
    const uint4* s4 = (const uint4*)wsrc;
    uint4* d4 = (uint4*)whts;
    for (int i = tid; i < (NPAD * KH_PAD) / 8; i += 256) d4[i] = s4[i];
  }
#endif
  for (int i = tid; i < 16 * KH_PAD; i += 256) hbf[i] = (__bf16)0.0f;

  // per-thread owned (row, j) elements of the 16 x 200 state
  const int NOWN = 13;                 // ceil(3200/256)
  float hreg[NOWN], sreg[NOWN], bz[NOWN], brr[NOWN], bhh[NOWN];
  int lenr[NOWN], rr_[NOWN], jj_[NOWN];
  #pragma unroll
  for (int e = 0; e < NOWN; ++e) {
    int idx = tid + e * 256;
    bool val = idx < 16 * HID;
    int r = val ? idx / HID : 0;
    int j = val ? idx % HID : 0;
    rr_[e] = r; jj_[e] = j;
    hreg[e] = 0.0f; sreg[e] = 0.0f;
    bz[e]  = val ? bh[j]           : 0.0f;
    brr[e] = val ? bh[HID + j]     : 0.0f;
    bhh[e] = val ? bh[2 * HID + j] : 0.0f;
    lenr[e] = val ? lengths[brow0 + r] : 0;
  }
  __syncthreads();

  for (int t = 0; t < TSEQ; ++t) {
    // --- rec = h @ Wh  (bf16 WMMA, f32 accumulate) ---
    v16bf areg[KH_CH];
    #pragma unroll
    for (int kc = 0; kc < KH_CH; ++kc)
      areg[kc] = load_a_frag(&hbf[lm * KH_PAD + kc * 32 + half * 8]);

    // prefetch next step's xp slice (38.4 KB) behind the WMMA work
    if (t + 1 < TSEQ) {
      const char* nb = (const char*)(xp + ((size_t)(t + 1) * BATCH + brow0) * H3);
      int off = tid * 128;
      __builtin_prefetch(nb + off, 0, 3);
      off += 32768;
      if (off < 16 * H3 * 4) __builtin_prefetch(nb + off, 0, 3);
    }

    for (int tile = wave; tile < NTILES; tile += 8) {
      v8f acc = (v8f){};
      #pragma unroll
      for (int kc = 0; kc < KH_CH; ++kc) {
        v16bf bm = load_a_frag(&whts[(size_t)(tile * 16 + lm) * KH_PAD
                                        + kc * 32 + half * 8]);
        acc = __builtin_amdgcn_wmma_f32_16x16x32_bf16(
            false, areg[kc], false, bm, (short)0, acc, false, false);
      }
      const int col = tile * 16 + lm;
      #pragma unroll
      for (int g = 0; g < 8; ++g)
        recs[(half * 8 + g) * NPAD + col] = acc[g];
    }
    __syncthreads();

    // --- gates: z,r,hh ; masked state update + masked sum ---
    const float* xpt = xp + ((size_t)t * BATCH + brow0) * H3;
    #pragma unroll
    for (int e = 0; e < NOWN; ++e) {
      int idx = tid + e * 256;
      if (idx < 16 * HID) {
        int r = rr_[e], j = jj_[e];
        const float* xr_ = xpt + (size_t)r * H3;
        float xz = xr_[j], xg = xr_[HID + j], xh = xr_[2 * HID + j];
        float rz = recs[r * NPAD + j]           + bz[e];
        float rg = recs[r * NPAD + HID + j]     + brr[e];
        float rh = recs[r * NPAD + 2 * HID + j] + bhh[e];
        float z   = sigmoidf_(xz + rz);
        float rgt = sigmoidf_(xg + rg);
        float hh  = fast_tanh(xh + rgt * rh);
        float hn  = z * hreg[e] + (1.0f - z) * hh;
        if (t < lenr[e]) { hreg[e] = hn; sreg[e] += hn; }
        hbf[r * KH_PAD + j] = (__bf16)hreg[e];
      }
    }
    __syncthreads();
  }

  #pragma unroll
  for (int e = 0; e < NOWN; ++e) {
    int idx = tid + e * 256;
    if (idx < 16 * HID)
      sums[((size_t)dir * BATCH + brow0 + rr_[e]) * HID + jj_[e]] = sreg[e];
  }
}

// ---------------- 4) pooled / L^2 -> dense -> tanh ----------------
__global__ void dense_tanh_k(const float* __restrict__ sums,
                             const int* __restrict__ lengths,
                             const float* __restrict__ Wd,
                             const float* __restrict__ bd,
                             float* __restrict__ out) {
  int b = blockIdx.x;
  int o = threadIdx.x;
  if (o >= OUTN) return;
  float L = (float)lengths[b];
  float inv = 1.0f / (L * L);
  float acc = bd[o];
  const float* sf = sums + (size_t)b * HID;
  const float* sb = sums + (size_t)(BATCH + b) * HID;
  for (int j = 0; j < HID; ++j) acc += sf[j] * inv * Wd[(size_t)j * OUTN + o];
  for (int j = 0; j < HID; ++j) acc += sb[j] * inv * Wd[(size_t)(HID + j) * OUTN + o];
  out[(size_t)b * OUTN + o] = fast_tanh(acc);
}

// ---------------- launch ----------------
extern "C" void kernel_launch(void* const* d_in, const int* in_sizes, int n_in,
                              void* d_out, int out_size, void* d_ws, size_t ws_size,
                              hipStream_t stream) {
  const float* x     = (const float*)d_in[0];
  const int*   lens  = (const int*)  d_in[1];
  const float* Wx_f  = (const float*)d_in[2];
  const float* Wh_f  = (const float*)d_in[3];
  const float* bx_f  = (const float*)d_in[4];
  const float* bh_f  = (const float*)d_in[5];
  const float* Wx_b  = (const float*)d_in[6];
  const float* Wh_b  = (const float*)d_in[7];
  const float* bx_b  = (const float*)d_in[8];
  const float* bh_b  = (const float*)d_in[9];
  const float* Wd    = (const float*)d_in[10];
  const float* bd    = (const float*)d_in[11];

  // workspace layout
  char* ws = (char*)d_ws;
  size_t off = 0;
  float* xp_f = (float*)(ws + off); off += (size_t)TSEQ * BATCH * H3 * 4;
  float* xp_b = (float*)(ws + off); off += (size_t)TSEQ * BATCH * H3 * 4;
  unsigned short* wxt = (unsigned short*)(ws + off); off += (size_t)2 * NPAD * KD_PAD * 2;
  unsigned short* wht = (unsigned short*)(ws + off); off += (size_t)2 * NPAD * KH_PAD * 2;
  float* sums = (float*)(ws + off);

  // 1) weight transpose/convert (tiny)
  {
    int totalx = NPAD * KD_PAD;
    int bl = (totalx + 255) / 256;
    transpose_pad_bf16_k<<<bl, 256, 0, stream>>>(Wx_f, wxt,                  DIN, H3, KD_PAD, NPAD);
    transpose_pad_bf16_k<<<bl, 256, 0, stream>>>(Wx_b, wxt + NPAD * KD_PAD,  DIN, H3, KD_PAD, NPAD);
    int totalh = NPAD * KH_PAD;
    bl = (totalh + 255) / 256;
    transpose_pad_bf16_k<<<bl, 256, 0, stream>>>(Wh_f, wht,                  HID, H3, KH_PAD, NPAD);
    transpose_pad_bf16_k<<<bl, 256, 0, stream>>>(Wh_b, wht + NPAD * KH_PAD,  HID, H3, KH_PAD, NPAD);
  }

  // 2) x projection GEMM (both directions; backward gathers reversed rows)
  dim3 g1((BATCH * TSEQ) / 64, (NPAD + 127) / 128, 2);
  xproj_wmma_k<<<g1, 256, 0, stream>>>(x, lens, wxt, bx_f, bx_b, xp_f, xp_b);

  // 3) sequential GRU scan, Wh^T resident in 320KB WGP LDS
  size_t lds = (size_t)(NPAD * KH_PAD + 16 * KH_PAD) * 2 + (size_t)16 * NPAD * 4;
  gru_scan_wmma_k<<<dim3(BATCH / 16, 2), 256, lds, stream>>>(
      xp_f, xp_b, wht, bh_f, bh_b, lens, sums);

  // 4) head
  dense_tanh_k<<<BATCH, 128, 0, stream>>>(sums, lens, Wd, bd, (float*)d_out);
}
